// MSPSurfNet_69638599737455
// MI455X (gfx1250) — matrix operand order repaired and checked by reading, not compiled, not run on hardware
//
#include <hip/hip_runtime.h>
#include <hip/hip_bf16.h>

// ---------------------------------------------------------------------------
// MSPSurfNet forward for MI455X (gfx1250, wave32, WMMA).
//
// Heavy stages (RBF projection w@feats, GCN An@X) run on
// v_wmma_f32_16x16x32_bf16 with the adjacency/weight matrix generated on the
// fly per-lane directly in A-fragment layout (contiguous k = 16*(lane/16) + s, same as B). The dense feature
// operand is register-double-buffered from global memory (loads for tile k+32
// issued while tile k computes) and committed to LDS transposed in bf16 so
// each lane's B-fragment is one contiguous 32B LDS load (2x ds_load_b128).
// bf16 packing uses v_perm_b32 (1 perm + 2 adds per pair).
// ---------------------------------------------------------------------------

typedef __attribute__((ext_vector_type(16))) __bf16    v16bf;
typedef __attribute__((ext_vector_type(8)))  float     v8f;
typedef __attribute__((ext_vector_type(8)))  unsigned  v8u;

#define NC 4096          // coords per side
#define CH 64            // feature channels
#define XT_STRIDE 40     // LDS row stride in elements (80B, 16B-aligned, bank-spread)

#define LOG2E 1.442695040888963f

// Pack two f32 into packed bf16 {lo=a, hi=b}: bias for round-half-up, then one
// v_perm_b32 grabs both high halves (sel bytes 0..3 <- src1, 4..7 <- src0).
__device__ __forceinline__ unsigned pk_u32(float a, float b) {
  unsigned ua = __builtin_bit_cast(unsigned, a) + 0x8000u;
  unsigned ub = __builtin_bit_cast(unsigned, b) + 0x8000u;
  return __builtin_amdgcn_perm(ub, ua, 0x07060302u);
}

// ---------------------------------------------------------------------------
// RBF projection: out[m, colOff + c] = (sum_k w(m,k) feats[k,c]) / sum_k w(m,k)
//                 out[m, colOff + 64] = min_k d(m,k),  w = exp(-d/2.5)
// 128 threads = 4 waves; each wave owns a 16-row M tile; K tiled by 32.
// K (=Nv=16000) must be a multiple of 32.
// ---------------------------------------------------------------------------
__global__ __launch_bounds__(128) void rbf_project_kernel(
    const float* __restrict__ verts,    // [K,3]
    const float* __restrict__ feats,    // [K,64]
    const float* __restrict__ coords,   // [NC,3]
    float* __restrict__ out,            // [NC,130]
    int K, int colOff)
{
  __shared__ __align__(16) __bf16 Xt[CH * XT_STRIDE];   // feats tile, transposed bf16
  __shared__ float Pts[96];
  __shared__ float Scale[128];
  __shared__ float Mind[128];

  const int tid  = threadIdx.x;
  const int lane = tid & 31;
  const int wave = tid >> 5;
  const int m0   = blockIdx.x * 64 + wave * 16;
  const int mr   = m0 + (lane & 15);

  const float cx = coords[mr * 3 + 0];
  const float cy = coords[mr * 3 + 1];
  const float cz = coords[mr * 3 + 2];

  const int  kbase = (lane < 16) ? 0 : 8;    // unused: A and B both take contiguous K half koff
  const int  koff  = (lane < 16) ? 0 : 16;   // B-fragment K half
  const float nls  = -LOG2E / 2.5f;          // exp(-d/s) == exp2(d * nls)

  // Staging slice: 8 consecutive cols at c0, two adjacent K rows kk0, kk0+1.
  // Consecutive threads sweep one 256B row -> coalesced; pairs pack to b32.
  const int c0  = (tid & 7) * 8;
  const int kk0 = (tid >> 3) * 2;

  v8f acc[4];
  #pragma unroll
  for (int n = 0; n < 4; ++n) acc[n] = (v8f){};
  float wsum = 0.0f;
  float dmin = 3.4e38f;

  // Preload tile 0 into registers (double buffer).
  float fr0[8], fr1[8];
  #pragma unroll
  for (int i = 0; i < 8; ++i) {
    fr0[i] = feats[kk0 * CH + c0 + i];
    fr1[i] = feats[(kk0 + 1) * CH + c0 + i];
  }
  float ptsR = (tid < 96) ? verts[tid] : 0.0f;

  for (int k0 = 0; k0 < K; k0 += 32) {
    // Commit staged registers -> LDS (transposed, packed bf16, b32 stores).
    #pragma unroll
    for (int i = 0; i < 8; ++i)
      *(unsigned*)(&Xt[(c0 + i) * XT_STRIDE + kk0]) = pk_u32(fr0[i], fr1[i]);
    if (tid < 96) Pts[tid] = ptsR;
    __syncthreads();

    // Issue global loads for tile k0+32 now; VMEM latency overlaps compute.
    const int k1 = k0 + 32;
    if (k1 < K) {
      #pragma unroll
      for (int i = 0; i < 8; ++i) {
        fr0[i] = feats[(k1 + kk0) * CH + c0 + i];
        fr1[i] = feats[(k1 + kk0 + 1) * CH + c0 + i];
      }
      if (tid < 96) ptsR = verts[k1 * 3 + tid];
      if (k1 + 32 < K)                       // warm L2 two tiles ahead
        __builtin_prefetch(&feats[(k1 + 32 + kk0) * CH + c0], 0, 1);
    }

    // Generate this lane's A-fragment (16 bf16 weights) directly in registers.
    float wv[16];
    #pragma unroll
    for (int t = 0; t < 16; ++t) {
      int kl = koff + t;
      float dx = cx - Pts[kl * 3 + 0];
      float dy = cy - Pts[kl * 3 + 1];
      float dz = cz - Pts[kl * 3 + 2];
      float d  = sqrtf(dx * dx + dy * dy + dz * dz);
      float w  = exp2f(d * nls);
      wsum += w;
      dmin  = fminf(dmin, d);
      wv[t] = w;
    }
    v8u av;
    #pragma unroll
    for (int t = 0; t < 8; ++t) av[t] = pk_u32(wv[2 * t], wv[2 * t + 1]);
    v16bf a = __builtin_bit_cast(v16bf, av);

    const __bf16* xb = &Xt[(lane & 15) * XT_STRIDE + koff];
    #pragma unroll
    for (int n = 0; n < 4; ++n) {
      v16bf b = *(const v16bf*)(xb + n * 16 * XT_STRIDE);
      acc[n] = __builtin_amdgcn_wmma_f32_16x16x32_bf16(
          false, a, false, b, (short)0, acc[n], false, false);
    }
    __syncthreads();
  }

  // Lanes l and l^16 cover complementary K halves of the same row -> pair-reduce.
  wsum += __shfl_xor(wsum, 16, 32);
  dmin  = fminf(dmin, __shfl_xor(dmin, 16, 32));
  if (lane < 16) {
    Scale[wave * 16 + lane] = 1.0f / wsum;
    Mind [wave * 16 + lane] = dmin;
  }
  __syncthreads();

  const int hi8 = (lane < 16) ? 0 : 8;
  #pragma unroll
  for (int n = 0; n < 4; ++n) {
    int col = colOff + n * 16 + (lane & 15);
    #pragma unroll
    for (int v = 0; v < 8; ++v) {
      int row = m0 + v + hi8;
      out[row * 130 + col] = acc[n][v] * Scale[wave * 16 + v + hi8];
    }
  }
  if (lane < 16)
    out[(m0 + lane) * 130 + colOff + CH] = Mind[wave * 16 + lane];
}

// ---------------------------------------------------------------------------
// dinv[i] = rsqrt( sum_j exp(-dist(i,j)/3) )   (includes diagonal exp(0)=1)
// ---------------------------------------------------------------------------
__global__ __launch_bounds__(128) void degree_kernel(
    const float* __restrict__ coords, float* __restrict__ dinv, int N)
{
  __shared__ float red[128];
  const int i = blockIdx.x;
  const float cx = coords[i * 3 + 0];
  const float cy = coords[i * 3 + 1];
  const float cz = coords[i * 3 + 2];
  const float nls = -LOG2E / 3.0f;
  float s = 0.0f;
  for (int j = threadIdx.x; j < N; j += 128) {
    float dx = cx - coords[j * 3 + 0];
    float dy = cy - coords[j * 3 + 1];
    float dz = cz - coords[j * 3 + 2];
    s += exp2f(sqrtf(dx * dx + dy * dy + dz * dz) * nls);
  }
  red[threadIdx.x] = s;
  __syncthreads();
  for (int o = 64; o > 0; o >>= 1) {
    if (threadIdx.x < o) red[threadIdx.x] += red[threadIdx.x + o];
    __syncthreads();
  }
  if (threadIdx.x == 0) dinv[i] = rsqrtf(red[0]);
}

// ---------------------------------------------------------------------------
// GCN adjacency apply: out = relu?( An @ Y + bias ), An recomputed tile-wise:
// An(i,j) = dinv[i] * exp(-dist(i,j)/3) * dinv[j].  Same WMMA scheme as RBF.
// ---------------------------------------------------------------------------
__global__ __launch_bounds__(128) void gcn_apply_kernel(
    const float* __restrict__ coords,   // [N,3]
    const float* __restrict__ dinv,     // [N]
    const float* __restrict__ Y,        // [N,64]
    const float* __restrict__ bias,     // [64]
    float* __restrict__ out,            // [N,64]
    int N, int doRelu)
{
  __shared__ __align__(16) __bf16 Xt[CH * XT_STRIDE];
  __shared__ float Pts[96];
  __shared__ float Dj[32];

  const int tid  = threadIdx.x;
  const int lane = tid & 31;
  const int wave = tid >> 5;
  const int m0   = blockIdx.x * 64 + wave * 16;
  const int mr   = m0 + (lane & 15);

  const float cx = coords[mr * 3 + 0];
  const float cy = coords[mr * 3 + 1];
  const float cz = coords[mr * 3 + 2];
  const float di = dinv[mr];
  const float nls = -LOG2E / 3.0f;

  const int kbase = (lane < 16) ? 0 : 8;
  const int koff  = (lane < 16) ? 0 : 16;

  const int c0  = (tid & 7) * 8;
  const int kk0 = (tid >> 3) * 2;

  v8f acc[4];
  #pragma unroll
  for (int n = 0; n < 4; ++n) acc[n] = (v8f){};

  float fr0[8], fr1[8];
  #pragma unroll
  for (int i = 0; i < 8; ++i) {
    fr0[i] = Y[kk0 * CH + c0 + i];
    fr1[i] = Y[(kk0 + 1) * CH + c0 + i];
  }
  float ptsR = (tid < 96) ? coords[tid] : 0.0f;
  float djR  = (tid < 32) ? dinv[tid] : 0.0f;

  for (int k0 = 0; k0 < N; k0 += 32) {
    #pragma unroll
    for (int i = 0; i < 8; ++i)
      *(unsigned*)(&Xt[(c0 + i) * XT_STRIDE + kk0]) = pk_u32(fr0[i], fr1[i]);
    if (tid < 96) Pts[tid] = ptsR;
    if (tid < 32) Dj[tid]  = djR;
    __syncthreads();

    const int k1 = k0 + 32;
    if (k1 < N) {
      #pragma unroll
      for (int i = 0; i < 8; ++i) {
        fr0[i] = Y[(k1 + kk0) * CH + c0 + i];
        fr1[i] = Y[(k1 + kk0 + 1) * CH + c0 + i];
      }
      if (tid < 96) ptsR = coords[k1 * 3 + tid];
      if (tid < 32) djR  = dinv[k1 + tid];
      if (k1 + 32 < N)
        __builtin_prefetch(&Y[(k1 + 32 + kk0) * CH + c0], 0, 1);
    }

    float wv[16];
    #pragma unroll
    for (int t = 0; t < 16; ++t) {
      int kl = koff + t;
      float dx = cx - Pts[kl * 3 + 0];
      float dy = cy - Pts[kl * 3 + 1];
      float dz = cz - Pts[kl * 3 + 2];
      float d  = sqrtf(dx * dx + dy * dy + dz * dz);
      wv[t] = exp2f(d * nls) * di * Dj[kl];
    }
    v8u av;
    #pragma unroll
    for (int t = 0; t < 8; ++t) av[t] = pk_u32(wv[2 * t], wv[2 * t + 1]);
    v16bf a = __builtin_bit_cast(v16bf, av);

    const __bf16* xb = &Xt[(lane & 15) * XT_STRIDE + koff];
    #pragma unroll
    for (int n = 0; n < 4; ++n) {
      v16bf b = *(const v16bf*)(xb + n * 16 * XT_STRIDE);
      acc[n] = __builtin_amdgcn_wmma_f32_16x16x32_bf16(
          false, a, false, b, (short)0, acc[n], false, false);
    }
    __syncthreads();
  }

  const int hi8 = (lane < 16) ? 0 : 8;
  #pragma unroll
  for (int n = 0; n < 4; ++n) {
    int col = n * 16 + (lane & 15);
    float bv = bias[col];
    #pragma unroll
    for (int v = 0; v < 8; ++v) {
      int row = m0 + v + hi8;
      float o = acc[n][v] + bv;
      if (doRelu) o = fmaxf(o, 0.0f);
      out[row * CH + col] = o;
    }
  }
}

// ---------------------------------------------------------------------------
// Small row-major GEMM: Y[M,N] = X[M,K] @ W[K,N]   (K <= 130, tiny vs rest)
// ---------------------------------------------------------------------------
__global__ void gemm_rm_kernel(const float* __restrict__ X,
                               const float* __restrict__ W,
                               float* __restrict__ Y, int M, int K, int N)
{
  int idx = blockIdx.x * blockDim.x + threadIdx.x;
  if (idx >= M * N) return;
  int r = idx / N, c = idx % N;
  float s = 0.0f;
  for (int k = 0; k < K; ++k) s = fmaf(X[r * K + k], W[k * N + c], s);
  Y[idx] = s;
}

// ---------------------------------------------------------------------------
// Column-wise max over N rows for two [N,64] buffers -> x[128]
// ---------------------------------------------------------------------------
__global__ void maxpool_kernel(const float* __restrict__ A,
                               const float* __restrict__ B,
                               float* __restrict__ x, int N)
{
  int t = threadIdx.x;                    // 128 threads
  const float* src = (t < CH) ? A : B;
  int c = t & (CH - 1);
  float m = -3.4e38f;
  for (int i = 0; i < N; ++i) m = fmaxf(m, src[i * CH + c]);
  x[t] = m;
}

// ---------------------------------------------------------------------------
// Final MLP: out = relu(x @ wm1 + bm1) @ wm2 + bm2
// ---------------------------------------------------------------------------
__global__ void mlp_kernel(const float* __restrict__ x,
                           const float* __restrict__ wm1,
                           const float* __restrict__ bm1,
                           const float* __restrict__ wm2,
                           const float* __restrict__ bm2,
                           float* __restrict__ out)
{
  __shared__ float xs[128];
  __shared__ float red[128];
  int t = threadIdx.x;
  xs[t] = x[t];
  __syncthreads();
  float h = bm1[t];
  for (int i = 0; i < 128; ++i) h = fmaf(xs[i], wm1[i * 128 + t], h);
  h = fmaxf(h, 0.0f);
  red[t] = h * wm2[t];
  __syncthreads();
  for (int o = 64; o > 0; o >>= 1) {
    if (t < o) red[t] += red[t + o];
    __syncthreads();
  }
  if (t == 0) out[0] = red[0] + bm2[0];
}

// ---------------------------------------------------------------------------
extern "C" void kernel_launch(void* const* d_in, const int* in_sizes, int n_in,
                              void* d_out, int out_size, void* d_ws, size_t ws_size,
                              hipStream_t stream) {
  (void)in_sizes; (void)n_in; (void)out_size; (void)ws_size;
  // setup_inputs() dict order: verts0,feats0,verts1,feats1,verts2,feats2,
  // verts3,feats3,coords0,coords1,w1,b1,w2,b2,w3,b3,wm1,bm1,wm2,bm2
  const float* verts[4] = {(const float*)d_in[0], (const float*)d_in[2],
                           (const float*)d_in[4], (const float*)d_in[6]};
  const float* feats[4] = {(const float*)d_in[1], (const float*)d_in[3],
                           (const float*)d_in[5], (const float*)d_in[7]};
  const float* coords0 = (const float*)d_in[8];
  const float* coords1 = (const float*)d_in[9];
  const float* w1 = (const float*)d_in[10]; const float* b1 = (const float*)d_in[11];
  const float* w2 = (const float*)d_in[12]; const float* b2 = (const float*)d_in[13];
  const float* w3 = (const float*)d_in[14]; const float* b3 = (const float*)d_in[15];
  const float* wm1 = (const float*)d_in[16]; const float* bm1 = (const float*)d_in[17];
  const float* wm2 = (const float*)d_in[18]; const float* bm2 = (const float*)d_in[19];
  float* out = (float*)d_out;

  const int Nc = NC, Nv = 16000;

  // Workspace layout (floats); total ~8.5 MB.
  float* ws     = (float*)d_ws;
  float* projO  = ws;                      // [4096,130]
  float* projM  = projO + Nc * 130;        // [4096,130]
  float* dinv0  = projM + Nc * 130;        // [4096]
  float* dinv1  = dinv0 + Nc;              // [4096]
  float* ybuf   = dinv1 + Nc;              // [4096,64]
  float* zbuf   = ybuf + Nc * CH;          // [4096,64]
  float* nodes0 = zbuf + Nc * CH;          // [4096,64]
  float* nodes1 = nodes0 + Nc * CH;        // [4096,64]
  float* xpool  = nodes1 + Nc * CH;        // [128]

  dim3 blk(128);
  dim3 grdM(Nc / 64);                      // 64 WGs, each 4 waves x 16 rows

  // 4 RBF projections -> concatenated [feats|min_d|feats|min_d] rows of 130.
  rbf_project_kernel<<<grdM, blk, 0, stream>>>(verts[0], feats[0], coords0, projO, Nv, 0);
  rbf_project_kernel<<<grdM, blk, 0, stream>>>(verts[1], feats[1], coords0, projO, Nv, 65);
  rbf_project_kernel<<<grdM, blk, 0, stream>>>(verts[2], feats[2], coords1, projM, Nv, 0);
  rbf_project_kernel<<<grdM, blk, 0, stream>>>(verts[3], feats[3], coords1, projM, Nv, 65);

  // Degree normalizers.
  degree_kernel<<<Nc, blk, 0, stream>>>(coords0, dinv0, Nc);
  degree_kernel<<<Nc, blk, 0, stream>>>(coords1, dinv1, Nc);

  const int gemmGrid = (Nc * CH + 255) / 256;
  auto run_gcn = [&](const float* coords, const float* dv, const float* proj, float* nodes) {
    gemm_rm_kernel<<<gemmGrid, 256, 0, stream>>>(proj, w1, ybuf, Nc, 130, CH);
    gcn_apply_kernel<<<grdM, blk, 0, stream>>>(coords, dv, ybuf, b1, zbuf, Nc, 1);
    gemm_rm_kernel<<<gemmGrid, 256, 0, stream>>>(zbuf, w2, ybuf, Nc, CH, CH);
    gcn_apply_kernel<<<grdM, blk, 0, stream>>>(coords, dv, ybuf, b2, zbuf, Nc, 1);
    gemm_rm_kernel<<<gemmGrid, 256, 0, stream>>>(zbuf, w3, ybuf, Nc, CH, CH);
    gcn_apply_kernel<<<grdM, blk, 0, stream>>>(coords, dv, ybuf, b3, nodes, Nc, 0);
  };
  run_gcn(coords0, dinv0, projO, nodes0);
  run_gcn(coords1, dinv1, projM, nodes1);

  maxpool_kernel<<<1, blk, 0, stream>>>(nodes0, nodes1, xpool, Nc);
  mlp_kernel<<<1, blk, 0, stream>>>(xpool, wm1, bm1, wm2, bm2, out);
}
